// QKVAttention_65369402245297
// MI455X (gfx1250) — matrix operand order repaired and checked by reading, not compiled, not run on hardware
//
#include <hip/hip_runtime.h>

// QKV attention, flash-style, f16 WMMA with fp32 accumulate + fp32 softmax.
// qkv: [4][1536][2048] f32 ; out: [4][512][2048] f32
// bs=4, H=8, C=64, T=2048, scale = 1/sqrt(64) = 0.125 (folded with log2(e))

#define T_SEQ 2048
#define CH    64
#define BT    64      // query rows per workgroup (16 per wave)
#define BKV   64      // key/value tile
#define NW    4       // waves per workgroup
#define LSTR  66      // f16 LDS row stride (33 banks, conflict-friendly, pairs aligned)

// KV double-buffer geometry (bytes)
#define KS_OFF   8448
#define VS_OFF   16896
#define BUF_STRIDE 16896   // KS1 = KS0 + stride, VS1 = VS0 + stride
#define PS_OFF   42240

// 0.125 * log2(e): softmax computed in base-2 domain -> bare v_exp_f32
#define QSCALE 0.180336879963407f

#if defined(__has_builtin) && __has_builtin(__builtin_amdgcn_exp2f)
#define EXP2F __builtin_amdgcn_exp2f
#else
#define EXP2F exp2f
#endif

typedef _Float16 v16h __attribute__((ext_vector_type(16)));
typedef float    v8f  __attribute__((ext_vector_type(8)));

union Frag16 {
    v16h v;
    unsigned u[8];
};

__device__ __forceinline__ unsigned ld_b32(const _Float16* p) {
    return *(const unsigned*)p;
}

__device__ __forceinline__ v8f wmma_f16(v16h a, v16h b, v8f c) {
    return __builtin_amdgcn_wmma_f32_16x16x32_f16(
        /*neg_a=*/false, a, /*neg_b=*/false, b,
        /*c_mod=*/(short)0, c, /*reuse_a=*/false, /*reuse_b=*/false);
}

// A-frag (16x32 f16) k-pair base for vgpr v, lane-half h
__device__ __forceinline__ int a_kb(int v, int h) {
    return 2 * (v & 3) + 8 * h + 16 * (v >> 2);
}

__global__ __launch_bounds__(NW * 32)
void qkv_attn_flash_kernel(const float* __restrict__ qkv, float* __restrict__ out) {
    // LDS carve: QS | KS0 | VS0 | KS1 | VS1 | PS   (50688 B total)
    __shared__ __align__(16) unsigned char smem[50688];
    _Float16* QS = (_Float16*)(smem);             // [BT][LSTR]  q*QSCALE, layout [t][c]
    _Float16* PS = (_Float16*)(smem + PS_OFF);    // [NW][16][LSTR] per-wave P staging

    const int tid  = threadIdx.x;
    const int wave = tid >> 5;
    const int lane = tid & 31;
    const int half = lane >> 4;
    const int l16  = lane & 15;

    const int bh = blockIdx.y;       // 0..31 = b*8 + h
    const int b  = bh >> 3;
    const int h  = bh & 7;
    const int t0 = blockIdx.x * BT;

    const float* qg = qkv + (size_t)(b * 1536 +        h * 64) * T_SEQ;
    const float* kg = qkv + (size_t)(b * 1536 +  512 + h * 64) * T_SEQ;
    const float* vg = qkv + (size_t)(b * 1536 + 1024 + h * 64) * T_SEQ;
    float*       og = out + (size_t)(b * 512  +        h * 64) * T_SEQ;

    // ---- Stage Q tile: global [c][t] (coalesced over t) -> LDS [t][c], pre-scaled
    for (int idx = tid; idx < CH * BT; idx += NW * 32) {
        int c = idx >> 6;
        int t = idx & 63;
        float qv = qg[(size_t)c * T_SEQ + (t0 + t)] * QSCALE;
        QS[t * LSTR + c] = (_Float16)qv;
    }
    __syncthreads();

    // ---- Persistent A-fragments of Q (this wave's 16 rows, k = c in [0,32),[32,64))
    Frag16 aq[2];
    const int arow = wave * 16 + l16;
#pragma unroll
    for (int kk = 0; kk < 2; ++kk)
#pragma unroll
        for (int v = 0; v < 8; ++v)
            aq[kk].u[v] = ld_b32(&QS[arow * LSTR + kk * 32 + a_kb(v, half)]);

    // ---- Prologue: stage KV tile 0 into buffer 0
    {
        _Float16* KS = (_Float16*)(smem + KS_OFF);
        _Float16* VS = (_Float16*)(smem + VS_OFF);
        for (int idx = tid; idx < CH * BKV; idx += NW * 32) {
            int c = idx >> 6;
            int s = idx & 63;
            KS[s * LSTR + c] = (_Float16)kg[(size_t)c * T_SEQ + s];
            VS[c * LSTR + s] = (_Float16)vg[(size_t)c * T_SEQ + s];
        }
    }

    // ---- Running softmax state (base-2 domain) + output accumulators
    float m_i[8], l_i[8];
    v8f   O[4];
    const v8f vzero = {0.f, 0.f, 0.f, 0.f, 0.f, 0.f, 0.f, 0.f};
#pragma unroll
    for (int r = 0; r < 8; ++r) { m_i[r] = -1e30f; l_i[r] = 0.f; }
#pragma unroll
    for (int j = 0; j < 4; ++j) O[j] = vzero;

    _Float16* PW = PS + wave * 16 * LSTR;

    // ================= KV loop (double-buffered, one barrier/iter) ===========
    for (int it = 0; it < T_SEQ / BKV; ++it) {
        const int p   = it & 1;
        const int s0n = (it + 1) * BKV;           // next tile base
        _Float16* KS = (_Float16*)(smem + KS_OFF + p * BUF_STRIDE);
        _Float16* VS = (_Float16*)(smem + VS_OFF + p * BUF_STRIDE);

        __syncthreads();   // buf[p] stores visible; buf[1-p] reads from it-1 done

        // Prefetch next KV tile into cache while we compute (no VGPR cost).
        if (s0n < T_SEQ) {
            int c  = tid >> 1;                    // 64 rows, 2 halves of 128B each
            int so = (tid & 1) * 32;
            __builtin_prefetch(&kg[(size_t)c * T_SEQ + s0n + so], 0, 1);
            __builtin_prefetch(&vg[(size_t)c * T_SEQ + s0n + so], 0, 1);
        }

        // ---- S(16xBKV) = Q^T K : 4 n-frags x 2 k-steps
        v8f S[4];
#pragma unroll
        for (int j = 0; j < 4; ++j) {
            Frag16 bk0, bk1;
#pragma unroll
            for (int v = 0; v < 8; ++v) {
                int kb = a_kb(v, half);           // B-frag: lane n, k pairs in vgprs
                const _Float16* base = &KS[(j * 16 + l16) * LSTR];
                bk0.u[v] = ld_b32(base + kb);
                bk1.u[v] = ld_b32(base + 32 + kb);
            }
            v8f acc = wmma_f16(aq[0].v, bk0.v, vzero);
            acc     = wmma_f16(aq[1].v, bk1.v, acc);
            S[j] = acc;
        }

        // ---- Online softmax in base 2 (row m = r + 8*half lives in this half)
#pragma unroll
        for (int r = 0; r < 8; ++r) {
            float mx = fmaxf(fmaxf(S[0][r], S[1][r]), fmaxf(S[2][r], S[3][r]));
            mx = fmaxf(mx, __shfl_xor(mx, 1, 32));
            mx = fmaxf(mx, __shfl_xor(mx, 2, 32));
            mx = fmaxf(mx, __shfl_xor(mx, 4, 32));
            mx = fmaxf(mx, __shfl_xor(mx, 8, 32));
            float mnew  = fmaxf(m_i[r], mx);
            float alpha = EXP2F(m_i[r] - mnew);
            float rs = 0.f;
#pragma unroll
            for (int j = 0; j < 4; ++j) {
                float pj = EXP2F(S[j][r] - mnew);
                S[j][r] = pj;
                rs += pj;
            }
            rs += __shfl_xor(rs, 1, 32);
            rs += __shfl_xor(rs, 2, 32);
            rs += __shfl_xor(rs, 4, 32);
            rs += __shfl_xor(rs, 8, 32);
            l_i[r] = l_i[r] * alpha + rs;
            m_i[r] = mnew;
#pragma unroll
            for (int j = 0; j < 4; ++j) O[j][r] *= alpha;
        }

        // ---- P (D-layout f32) -> per-wave LDS -> A-layout f16 frags
#pragma unroll
        for (int j = 0; j < 4; ++j)
#pragma unroll
            for (int r = 0; r < 8; ++r)
                PW[(r + 8 * half) * LSTR + j * 16 + l16] = (_Float16)S[j][r];

        Frag16 ap[2];
#pragma unroll
        for (int kk = 0; kk < 2; ++kk)
#pragma unroll
            for (int v = 0; v < 8; ++v)
                ap[kk].u[v] = ld_b32(&PW[l16 * LSTR + kk * 32 + a_kb(v, half)]);

        // ---- O += P V^T : n = c groups, k = s
#pragma unroll
        for (int jc = 0; jc < 4; ++jc) {
            Frag16 bv0, bv1;
#pragma unroll
            for (int v = 0; v < 8; ++v) {
                int kb = a_kb(v, half);
                const _Float16* base = &VS[(jc * 16 + l16) * LSTR];
                bv0.u[v] = ld_b32(base + kb);
                bv1.u[v] = ld_b32(base + 32 + kb);
            }
            O[jc] = wmma_f16(ap[0].v, bv0.v, O[jc]);
            O[jc] = wmma_f16(ap[1].v, bv1.v, O[jc]);
        }

        // ---- Stage next KV tile into the other buffer (prefetched above)
        if (s0n < T_SEQ) {
            _Float16* KSn = (_Float16*)(smem + KS_OFF + (p ^ 1) * BUF_STRIDE);
            _Float16* VSn = (_Float16*)(smem + VS_OFF + (p ^ 1) * BUF_STRIDE);
            for (int idx = tid; idx < CH * BKV; idx += NW * 32) {
                int c = idx >> 6;
                int s = idx & 63;
                KSn[s * LSTR + c] = (_Float16)kg[(size_t)c * T_SEQ + s0n + s];
                VSn[c * LSTR + s] = (_Float16)vg[(size_t)c * T_SEQ + s0n + s];
            }
        }
    }

    // ================= Epilogue: O/l, transpose via LDS, coalesced store ======
    __syncthreads();                       // all waves done; last tile used buf1
    float* OS = (float*)smem;              // [CH][BT] = 16 KB, overlays QS+KS0
#pragma unroll
    for (int r = 0; r < 8; ++r) {
        float inv = 1.0f / l_i[r];
        int trow = wave * 16 + r + 8 * half;
#pragma unroll
        for (int jc = 0; jc < 4; ++jc) {
            int c = jc * 16 + l16;
            OS[c * BT + trow] = O[jc][r] * inv;
        }
    }
    __syncthreads();
    for (int idx = tid; idx < CH * BT; idx += NW * 32) {
        int c = idx >> 6;
        int t = idx & 63;
        og[(size_t)c * T_SEQ + (t0 + t)] = OS[idx];   // 256B contiguous per row
    }
}

extern "C" void kernel_launch(void* const* d_in, const int* in_sizes, int n_in,
                              void* d_out, int out_size, void* d_ws, size_t ws_size,
                              hipStream_t stream) {
    (void)in_sizes; (void)n_in; (void)out_size; (void)d_ws; (void)ws_size;
    const float* qkv = (const float*)d_in[0];
    float* out = (float*)d_out;
    dim3 grid(T_SEQ / BT, 4 * 8);   // 32 query tiles x 32 batch-heads
    dim3 block(NW * 32);
    qkv_attn_flash_kernel<<<grid, block, 0, stream>>>(qkv, out);
}
